// LSTM_78176994722094
// MI455X (gfx1250) — matrix operand order repaired and checked by reading, compiled, not run on hardware
//
#include <hip/hip_runtime.h>
#include <hip/hip_bf16.h>

// ---------------------------------------------------------------------------
// 2-layer LSTM (B=128, D_IN=128, T=1024, H=512) + FC head on gfx1250.
//
//   Gate-permuted packing: packed weight row p <-> original gate row
//   q*512 + w*4 + u  (w = p>>4, q = (p>>2)&3, u = p&3), so workgroup w's 16
//   output columns are the full (i,f,g,o) quadruples for h-cols 4w..4w+3.
//   =>  pointwise LSTM cell is workgroup-local: gates stay in LDS, cell
//   state c stays in registers, and only ONE grid barrier per timestep is
//   needed (h double-buffered by timestep parity to avoid WAR).
//
//   128 WGs x 256 thr: wave wid = M-tile (16 batch rows), all waves share the
//   WG's 16 weight columns, which live in LDS (<=32KB, fragment-swizzled,
//   conflict-free ds_load_b128) for the entire layer. A (activations) streams
//   from L2. v_wmma_f32_16x16x32_bf16 with two interleaved accumulators.
//   K-loop split into input-part and h-part so no per-iteration selects.
// ---------------------------------------------------------------------------

typedef __attribute__((ext_vector_type(4)))  float  v4f;
typedef __attribute__((ext_vector_type(8)))  float  v8f;
typedef __attribute__((ext_vector_type(8)))  __bf16 v8bf;
typedef __attribute__((ext_vector_type(16))) __bf16 v16bf;

#define LSTM_B    128
#define LSTM_D    128
#define LSTM_T    1024
#define LSTM_H    512
#define LSTM_G    2048   // 4*H
#define NWG       128
#define NTHR      256

// packed row p -> original gate row
__host__ __device__ __forceinline__ int lstm_perm_row(int p) {
    int w = p >> 4, q = (p >> 2) & 3, u = p & 3;
    return q * LSTM_H + w * 4 + u;
}

// ------------------------------ prep kernels -------------------------------

__global__ void lstm_k_init(unsigned* bar) {
    bar[0] = 0u;  // count
    bar[1] = 0u;  // generation
}

// x (B, D, T) f32  ->  xT (T, B, D) bf16.  Whole x (64 MB) fits in 192 MB L2.
__global__ void lstm_k_xpose(const float* __restrict__ x, __bf16* __restrict__ xT) {
    size_t i = (size_t)blockIdx.x * blockDim.x + threadIdx.x;   // T*B*D = 16.7M
    int d = (int)(i & (LSTM_D - 1));
    size_t r = i >> 7;
    int b = (int)(r & (LSTM_B - 1));
    int t = (int)(r >> 7);
    xT[i] = (__bf16)x[((size_t)b * LSTM_D + d) * LSTM_T + t];
}

// W0p (2048 x 640) = permuted( [Wih0 | Whh0] ) bf16
__global__ void lstm_k_packw0(const float* __restrict__ Wih, const float* __restrict__ Whh,
                              __bf16* __restrict__ W0) {
    int i = blockIdx.x * blockDim.x + threadIdx.x;  // 2048*640
    int k = i % 640, p = i / 640;
    int g = lstm_perm_row(p);
    float v = (k < LSTM_D) ? Wih[g * LSTM_D + k] : Whh[g * LSTM_H + (k - LSTM_D)];
    W0[i] = (__bf16)v;
}

// W1p (2048 x 1024) = permuted( [Wih1 | Whh1] ) bf16
__global__ void lstm_k_packw1(const float* __restrict__ Wih, const float* __restrict__ Whh,
                              __bf16* __restrict__ W1) {
    int i = blockIdx.x * blockDim.x + threadIdx.x;  // 2048*1024
    int k = i & 1023, p = i >> 10;
    int g = lstm_perm_row(p);
    float v = (k < LSTM_H) ? Wih[g * LSTM_H + k] : Whh[g * LSTM_H + (k - LSTM_H)];
    W1[i] = (__bf16)v;
}

__global__ void lstm_k_bias(const float* bih0, const float* bhh0,
                            const float* bih1, const float* bhh1,
                            float* bias0, float* bias1) {
    int p = blockIdx.x * blockDim.x + threadIdx.x;  // 2048 (permuted index)
    int g = lstm_perm_row(p);
    bias0[p] = bih0[g] + bhh0[g];
    bias1[p] = bih1[g] + bhh1[g];
}

// ---------------------------- persistent kernel ----------------------------

__device__ __forceinline__ void grid_barrier(unsigned* cnt, unsigned* gen) {
    __syncthreads();
    if (threadIdx.x == 0) {
        unsigned g = __hip_atomic_load(gen, __ATOMIC_RELAXED, __HIP_MEMORY_SCOPE_AGENT);
        __threadfence();
        if (__hip_atomic_fetch_add(cnt, 1u, __ATOMIC_ACQ_REL, __HIP_MEMORY_SCOPE_AGENT)
            == NWG - 1) {
            __hip_atomic_store(cnt, 0u, __ATOMIC_RELAXED, __HIP_MEMORY_SCOPE_AGENT);
            __hip_atomic_fetch_add(gen, 1u, __ATOMIC_RELEASE, __HIP_MEMORY_SCOPE_AGENT);
        } else {
            while (__hip_atomic_load(gen, __ATOMIC_ACQUIRE, __HIP_MEMORY_SCOPE_AGENT) == g)
                __builtin_amdgcn_s_sleep(2);
        }
    }
    __syncthreads();
}

// A fragment (16x32 bf16), §7.12.2: lane<16 holds row M=lane, K {k0..k0+7, k0+16..k0+23};
// lane>=16 holds row M=lane-16, K {k0+8..k0+15, k0+24..k0+31}.
__device__ __forceinline__ v16bf load_afrag(const __bf16* rowp, int k0, int sel) {
    v8bf lo = *(const v8bf*)(rowp + k0 + sel * 8);
    v8bf hi = *(const v8bf*)(rowp + k0 + 16 + sel * 8);
    return __builtin_shufflevector(lo, hi, 0, 1, 2, 3, 4, 5, 6, 7,
                                   8, 9, 10, 11, 12, 13, 14, 15);
}

// B fragment ks from the LDS cache (lw = ldsW + lane*8)
__device__ __forceinline__ v16bf load_bfrag(const __bf16* lw, int ks) {
    v8bf lo = *(const v8bf*)(lw + ks * 512);
    v8bf hi = *(const v8bf*)(lw + ks * 512 + 256);
    return __builtin_shufflevector(lo, hi, 0, 1, 2, 3, 4, 5, 6, 7,
                                   8, 9, 10, 11, 12, 13, 14, 15);
}

__device__ __forceinline__ float sigmoidf_(float x) { return 1.0f / (1.0f + __expf(-x)); }

#define GPITCH 132          // padded f32 pitch for LDS gate tile (bank spread)

__global__ void __launch_bounds__(NTHR)
lstm_persistent(const __bf16* __restrict__ xT,   // (T,B,D) bf16
                __bf16* __restrict__ hs0,        // (T,B,H) bf16 (layer-0 outputs)
                const __bf16* __restrict__ W0,   // (2048 x 640)  permuted
                const __bf16* __restrict__ W1,   // (2048 x 1024) permuted
                const float* __restrict__ bias0, // (2048) permuted
                const float* __restrict__ bias1,
                __bf16* __restrict__ hst,        // 2 x (128 x 512) bf16, ping-pong
                unsigned* __restrict__ bar) {
    // B-fragment cache: frag ks occupies 512 bf16 (1KB): lane l's low 8
    // K-values at ks*512 + l*8, high 8 at ks*512 + 256 + l*8.
    __shared__ __bf16 ldsW[32 * 512];            // 32 KB  (KS <= 32)
    __shared__ float  ldsG[16 * GPITCH];         // gate tile [col][row], 8.25 KB

    const int wid  = threadIdx.x >> 5;           // 0..7  (M tile)
    const int lane = threadIdx.x & 31;
    const int l15  = lane & 15;
    const int sel  = lane >> 4;
    const int m0   = wid * 16;
    const int nb   = blockIdx.x * 16;            // this WG's 16 permuted gate cols
    const int jb   = blockIdx.x * 4;             // this WG's 4 h-columns
    const int arow = m0 + l15;                   // A-matrix row for this lane

    // pointwise element mapping: idx = threadIdx.x + e*NTHR -> (b, u)
    const int pw_u  = threadIdx.x & 3;
    const int pw_b0 = threadIdx.x >> 2;          // e=0: rows 0..63, e=1: 64..127

    for (int layer = 0; layer < 2; ++layer) {
        const __bf16* Wl   = layer ? W1 : W0;
        const float*  bl   = layer ? bias1 : bias0;
        const int     Dl   = layer ? LSTM_H : LSTM_D;     // input-part K
        const int     ldw  = Dl + LSTM_H;                 // total K
        const int     KS   = ldw >> 5;                    // k-steps (20 or 32)
        const int     ksIn = Dl >> 5;                     // input k-steps (4 or 16)

        // ---- stage this WG's 16 weight columns into LDS, fragment-swizzled ----
        for (int i = threadIdx.x; i < KS * 32; i += NTHR) {
            const int ln = i & 31;
            const int ks = i >> 5;
            const __bf16* src = Wl + (size_t)(nb + (ln & 15)) * ldw
                                   + ks * 32 + (ln >> 4) * 8;
            *(v8bf*)(ldsW + ks * 512 + ln * 8)       = *(const v8bf*)src;
            *(v8bf*)(ldsW + ks * 512 + 256 + ln * 8) = *(const v8bf*)(src + 16);
        }

        // ---- per-thread bias cache, zero cell state + h read buffer (t=0) ----
        float cr[2];
        float bI[2], bF[2], bG[2], bO[2];
#pragma unroll
        for (int e = 0; e < 2; ++e) {
            cr[e] = 0.0f;
            bI[e] = bl[nb + 0 * 4 + pw_u];
            bF[e] = bl[nb + 1 * 4 + pw_u];
            bG[e] = bl[nb + 2 * 4 + pw_u];
            bO[e] = bl[nb + 3 * 4 + pw_u];
            const int b = pw_b0 + e * 64;
            hst[b * LSTM_H + jb + pw_u] = (__bf16)0.0f;   // buffer 0 (= t=0 read)
        }
        grid_barrier(bar, bar + 1);

        const __bf16* inb  = layer ? hs0 : xT;
        const int     inst = LSTM_B * Dl;                 // per-t input stride

        for (int t = 0; t < LSTM_T; ++t) {
            const __bf16* in_t = inb + (size_t)t * inst;
            const __bf16* hR = hst + (size_t)(t & 1) * (LSTM_B * LSTM_H);
            __bf16*       hW = hst + (size_t)((t + 1) & 1) * (LSTM_B * LSTM_H);
            if (t + 1 < LSTM_T)
                __builtin_prefetch(in_t + inst + (size_t)arow * Dl, 0, 1);

            // ---- GEMM: 16x16 tile (rows m0.., cols nb..) over K = Dl+512 ----
            v8f accA = {}, accB = {};
            const __bf16* arow_in = in_t + (size_t)arow * Dl;
            const __bf16* arow_h  = hR + (size_t)arow * LSTM_H;
            const __bf16* lw = ldsW + lane * 8;

            // input part: A from in_t, fragments 0..ksIn-1 (ksIn is even)
            for (int ks = 0; ks < ksIn; ks += 2) {
                v16bf a0 = load_afrag(arow_in, ks * 32, sel);
                accA = __builtin_amdgcn_wmma_f32_16x16x32_bf16(
                    false, a0, false, load_bfrag(lw, ks), (short)0, accA, false, false);
                v16bf a1 = load_afrag(arow_in, ks * 32 + 32, sel);
                accB = __builtin_amdgcn_wmma_f32_16x16x32_bf16(
                    false, a1, false, load_bfrag(lw, ks + 1), (short)0, accB, false, false);
            }
            // h part: A from hR, fragments ksIn..ksIn+15 (16 steps, fixed)
            const __bf16* lwh = lw + ksIn * 512;
#pragma unroll 8
            for (int ks = 0; ks < 16; ks += 2) {
                v16bf a0 = load_afrag(arow_h, ks * 32, sel);
                accA = __builtin_amdgcn_wmma_f32_16x16x32_bf16(
                    false, a0, false, load_bfrag(lwh, ks), (short)0, accA, false, false);
                v16bf a1 = load_afrag(arow_h, ks * 32 + 32, sel);
                accB = __builtin_amdgcn_wmma_f32_16x16x32_bf16(
                    false, a1, false, load_bfrag(lwh, ks + 1), (short)0, accB, false, false);
            }
            v8f acc = accA + accB;

            // C/D layout: lane holds col nb+l15, rows m0+8*sel+0..7 (contiguous)
            {
                float* g0 = &ldsG[l15 * GPITCH + m0 + sel * 8];
                *(v4f*)g0       = __builtin_shufflevector(acc, acc, 0, 1, 2, 3);
                *(v4f*)(g0 + 4) = __builtin_shufflevector(acc, acc, 4, 5, 6, 7);
            }
            __syncthreads();

            // ---- WG-local pointwise cell: 128 b x 4 j over 256 threads ----
#pragma unroll
            for (int e = 0; e < 2; ++e) {
                const int b = pw_b0 + e * 64;
                const float gi = sigmoidf_(ldsG[(0 * 4 + pw_u) * GPITCH + b] + bI[e]);
                const float gf = sigmoidf_(ldsG[(1 * 4 + pw_u) * GPITCH + b] + bF[e]);
                const float gg = tanhf   (ldsG[(2 * 4 + pw_u) * GPITCH + b] + bG[e]);
                const float go = sigmoidf_(ldsG[(3 * 4 + pw_u) * GPITCH + b] + bO[e]);
                const float c  = gf * cr[e] + gi * gg;
                const float h  = go * tanhf(c);
                cr[e] = c;
                const __bf16 hb = (__bf16)h;
                hW[b * LSTM_H + jb + pw_u] = hb;
                if (layer == 0)
                    hs0[(size_t)t * (LSTM_B * LSTM_H) + b * LSTM_H + jb + pw_u] = hb;
            }
            grid_barrier(bar, bar + 1);   // h complete before next step's GEMM
        }
    }
    // after t=1023, the final h lives in buffer 0 (== hst base)
}

// ------------------------------- head kernel -------------------------------

__global__ void lstm_k_head(const __bf16* __restrict__ h,  // (128 x 512) h_last
                            const float* __restrict__ fcW, // (2 x 512)
                            const float* __restrict__ fcb, // (2)
                            float* __restrict__ out) {     // (128 x 2)
    const int i = threadIdx.x;            // 256 = 128*2
    const int b = i >> 1, o = i & 1;
    float s = fcb[o];
    for (int k = 0; k < LSTM_H; ++k)
        s += (float)h[b * LSTM_H + k] * fcW[o * LSTM_H + k];
    out[b * 2 + o] = s;
}

// ------------------------------ host launcher ------------------------------

extern "C" void kernel_launch(void* const* d_in, const int* in_sizes, int n_in,
                              void* d_out, int out_size, void* d_ws, size_t ws_size,
                              hipStream_t stream) {
    const float* x    = (const float*)d_in[0];
    const float* Wih0 = (const float*)d_in[1];
    const float* Whh0 = (const float*)d_in[2];
    const float* bih0 = (const float*)d_in[3];
    const float* bhh0 = (const float*)d_in[4];
    const float* Wih1 = (const float*)d_in[5];
    const float* Whh1 = (const float*)d_in[6];
    const float* bih1 = (const float*)d_in[7];
    const float* bhh1 = (const float*)d_in[8];
    const float* fcW  = (const float*)d_in[9];
    const float* fcb  = (const float*)d_in[10];
    float* out = (float*)d_out;

    // workspace carve (256B aligned slots)
    char* p = (char*)d_ws;
    auto take = [&](size_t bytes) {
        char* r = p;
        p += (bytes + 255) & ~(size_t)255;
        return r;
    };
    __bf16*   xT    = (__bf16*)take((size_t)LSTM_T * LSTM_B * LSTM_D * 2); // 32 MB
    __bf16*   hs0   = (__bf16*)take((size_t)LSTM_T * LSTM_B * LSTM_H * 2); // 128 MB
    __bf16*   W0    = (__bf16*)take((size_t)LSTM_G * (LSTM_D + LSTM_H) * 2);
    __bf16*   W1    = (__bf16*)take((size_t)LSTM_G * (2 * LSTM_H) * 2);
    float*    bias0 = (float*)take(LSTM_G * 4);
    float*    bias1 = (float*)take(LSTM_G * 4);
    __bf16*   hst   = (__bf16*)take((size_t)2 * LSTM_B * LSTM_H * 2);      // ping-pong
    unsigned* bar   = (unsigned*)take(256);

    lstm_k_init<<<1, 1, 0, stream>>>(bar);
    lstm_k_xpose<<<(LSTM_T * LSTM_B * LSTM_D) / 256, 256, 0, stream>>>(x, xT);
    lstm_k_packw0<<<(LSTM_G * (LSTM_D + LSTM_H)) / 256, 256, 0, stream>>>(Wih0, Whh0, W0);
    lstm_k_packw1<<<(LSTM_G * 2 * LSTM_H) / 256, 256, 0, stream>>>(Wih1, Whh1, W1);
    lstm_k_bias<<<LSTM_G / 256, 256, 0, stream>>>(bih0, bhh0, bih1, bhh1, bias0, bias1);

    lstm_persistent<<<NWG, NTHR, 0, stream>>>(xT, hs0, W0, W1, bias0, bias1, hst, bar);

    lstm_k_head<<<1, 256, 0, stream>>>(hst, fcW, fcb, out);
}
